// ImageEncoderViT_19121194401965
// MI455X (gfx1250) — matrix operand order repaired and checked by reading, compiled, not run on hardware
//
#include <hip/hip_runtime.h>
#include <math.h>

// ---------------------------------------------------------------------------
// Types for CDNA5 WMMA (wave32): v_wmma_f32_16x16x32_bf16
// ---------------------------------------------------------------------------
typedef __bf16 v16bf __attribute__((ext_vector_type(16)));
typedef __bf16 v4bf  __attribute__((ext_vector_type(4)));
typedef __bf16 v2bf  __attribute__((ext_vector_type(2)));
typedef float  v8f   __attribute__((ext_vector_type(8)));

union FragU {
    v16bf v;
    unsigned int u[8];
    unsigned short h[16];
};

// Native conversions: v_cvt_bf16_f32 / v_cvt_pk_bf16_f32
__device__ inline unsigned short f2bf(float f) {
    union { __bf16 b; unsigned short u; } r;
    r.b = (__bf16)f;
    return r.u;
}
__device__ inline unsigned int pack2(float lo, float hi) {
    v2bf t = { (__bf16)lo, (__bf16)hi };
    return __builtin_bit_cast(unsigned int, t);
}
// float4 -> 4 packed bf16 (8 bytes), for single ds_store_b64 staging
__device__ inline uint2 pack4(float4 v) {
    v4bf t = { (__bf16)v.x, (__bf16)v.y, (__bf16)v.z, (__bf16)v.w };
    return __builtin_bit_cast(uint2, t);
}

// A-operand fragment (16x32, 16-bit) from LDS tile [row][k], given row stride
// (elements) and k-offset. lanes 0-15: row=row0+lane, K {0..7,16..23};
// lanes 16-31: K {8..15,24..31}.
__device__ inline v16bf frag_a_ld(const unsigned short* lds, int row0,
                                  int stride, int kc) {
    int lane = threadIdx.x & 31;
    const unsigned short* p =
        lds + (row0 + (lane & 15)) * stride + kc + (lane >> 4) * 8;
    FragU f;
#pragma unroll
    for (int j = 0; j < 4; j++) f.u[j]     = *(const unsigned int*)(p + 2 * j);
#pragma unroll
    for (int j = 0; j < 4; j++) f.u[4 + j] = *(const unsigned int*)(p + 16 + 2 * j);
    return f.v;
}
// B-operand fragment (32x16, 16-bit) from LDS tile holding B^T [n][k].
// lanes 0-15: n=row0+lane, K {0..7,16..23}; lanes 16-31: K {8..15,24..31}
__device__ inline v16bf frag_b_ld(const unsigned short* lds, int row0,
                                  int stride, int kc) {
    int lane = threadIdx.x & 31;
    const unsigned short* p =
        lds + (row0 + (lane & 15)) * stride + kc + (lane >> 4) * 8;
    FragU f;
#pragma unroll
    for (int j = 0; j < 4; j++) f.u[j]     = *(const unsigned int*)(p + 2 * j);
#pragma unroll
    for (int j = 0; j < 4; j++) f.u[4 + j] = *(const unsigned int*)(p + 16 + 2 * j);
    return f.v;
}

// ---------------------------------------------------------------------------
// Generic WMMA GEMM:  C[M,N] = A[M,K] @ B  (+bias) (gelu) (+resid)
// A row-major fp32 (lda), B addressed as B[k*sBk + n*sBn] fp32.
// Requires M%128==0, N%128==0, K%32==0. Block=256 thr, tile 128x128.
// ---------------------------------------------------------------------------
template<bool BIAS, bool RESID, bool GELU_>
__global__ void __launch_bounds__(256)
gemm_kernel(const float* __restrict__ A, int lda,
            const float* __restrict__ B, long long sBk, long long sBn,
            const float* __restrict__ bias, const float* __restrict__ resid,
            float* __restrict__ C, int M, int N, int K)
{
    __shared__ __align__(16) unsigned short lA[128 * 36];
    __shared__ __align__(16) unsigned short lB[128 * 36];
    const int tid  = threadIdx.x;
    const int lane = tid & 31;
    const int wave = tid >> 5;
    const int wm = wave >> 2;          // 0..1 (M dir)
    const int wn = wave & 3;           // 0..3 (N dir)
    const int m0 = blockIdx.y * 128;
    const int n0 = blockIdx.x * 128;

    const v8f vzero = {0.f, 0.f, 0.f, 0.f, 0.f, 0.f, 0.f, 0.f};
    v8f acc[4][2];
#pragma unroll
    for (int a = 0; a < 4; a++)
#pragma unroll
        for (int b2 = 0; b2 < 2; b2++) acc[a][b2] = vzero;

    for (int kc = 0; kc < K; kc += 32) {
        // ---- Stage A tile [128m][32k]: float4 loads, single b64 LDS stores
#pragma unroll
        for (int i = 0; i < 4; i++) {
            int idx = i * 256 + tid;          // 0..1023
            int r = idx >> 3, c4 = (idx & 7) * 4;
            float4 av = *(const float4*)(A + (long long)(m0 + r) * lda + kc + c4);
            *(uint2*)&lA[r * 36 + c4] = pack4(av);
        }
        // ---- Stage B^T tile [128n][32k]
        if (sBn == 1) {                        // row-major weights [K,N]
#pragma unroll
            for (int i = 0; i < 4; i++) {
                int idx = i * 256 + tid;
                int kk = idx >> 5, n4 = (idx & 31) * 4;
                float4 bv = *(const float4*)(B + (long long)(kc + kk) * sBk + (n0 + n4));
                lB[(n4 + 0) * 36 + kk] = f2bf(bv.x);
                lB[(n4 + 1) * 36 + kk] = f2bf(bv.y);
                lB[(n4 + 2) * 36 + kk] = f2bf(bv.z);
                lB[(n4 + 3) * 36 + kk] = f2bf(bv.w);
            }
        } else if (sBk == 1) {                 // [N,K]-style weights (convs)
#pragma unroll
            for (int i = 0; i < 4; i++) {
                int idx = i * 256 + tid;
                int nn = idx >> 3, k4 = (idx & 7) * 4;
                float4 bv = *(const float4*)(B + (long long)(n0 + nn) * sBn + kc + k4);
                *(uint2*)&lB[nn * 36 + k4] = pack4(bv);
            }
        } else {                               // generic fallback
#pragma unroll
            for (int i = 0; i < 16; i++) {
                int idx = i * 256 + tid;
                int nn = idx & 127, kk = idx >> 7;
                lB[nn * 36 + kk] =
                    f2bf(B[(long long)(kc + kk) * sBk + (long long)(n0 + nn) * sBn]);
            }
        }
        __syncthreads();
#pragma unroll
        for (int mi = 0; mi < 4; mi++) {
            v16bf af = frag_a_ld(lA, wm * 64 + mi * 16, 36, 0);
#pragma unroll
            for (int ni = 0; ni < 2; ni++) {
                v16bf bfv = frag_b_ld(lB, wn * 32 + ni * 16, 36, 0);
                acc[mi][ni] = __builtin_amdgcn_wmma_f32_16x16x32_bf16(
                    false, af, false, bfv, (short)0, acc[mi][ni], false, false);
            }
        }
        __syncthreads();
    }

#pragma unroll
    for (int mi = 0; mi < 4; mi++) {
#pragma unroll
        for (int ni = 0; ni < 2; ni++) {
            int col  = n0 + wn * 32 + ni * 16 + (lane & 15);
            int rowb = m0 + wm * 64 + mi * 16 + (lane >> 4) * 8;
            float bv = BIAS ? bias[col] : 0.f;
#pragma unroll
            for (int r = 0; r < 8; r++) {
                float v = acc[mi][ni][r] + bv;
                if (GELU_) v = 0.5f * v * (1.f + erff(v * 0.70710678118654752f));
                if (RESID) v += resid[(long long)(rowb + r) * N + col];
                C[(long long)(rowb + r) * N + col] = v;
            }
        }
    }
}

// ---------------------------------------------------------------------------
// Row pointer for window-local token ti in window w: real tokens point into
// the qkv buffer; padded tokens (outside the 32x32 grid) point at the qkv
// bias row (== Linear of a zero row), matching the reference's zero-padding.
// Compile-time WS/WPS -> cheap div/mod.
// ---------------------------------------------------------------------------
template<int WS, int WPS>
__device__ inline const float* qkv_row(const float* __restrict__ qkv,
                                       const float* __restrict__ qb,
                                       int ti, int w)
{
    int wy = w / WPS, wx = w % WPS;
    int iy = ti / WS, ix = ti % WS;
    int y = wy * WS + iy, x = wx * WS + ix;
    return (y < 32 && x < 32) ? (qkv + (long long)(y * 32 + x) * 2304) : qb;
}

// ---------------------------------------------------------------------------
// Dh[q,j] = q_vec . rel_h[j],  Dw likewise (decomposed rel-pos projections)
// ---------------------------------------------------------------------------
template<bool GLOB>
__global__ void __launch_bounds__(256)
relq_kernel(const float* __restrict__ qkv, const float* __restrict__ qb,
            const float* __restrict__ relh, const float* __restrict__ relw,
            float* __restrict__ Dh, float* __restrict__ Dw)
{
    constexpr int T   = GLOB ? 1024 : 196;
    constexpr int TP  = GLOB ? 1024 : 256;
    constexpr int WS  = GLOB ? 32 : 14;
    constexpr int WPS = GLOB ? 1 : 3;
    constexpr int SR  = GLOB ? 32 : 14;
    constexpr int NREL = 2 * SR - 1;
    const int wh = blockIdx.y;
    const int w = wh / 12, n = wh % 12;
    float* DhP = Dh + (long long)wh * TP * 64;
    float* DwP = Dw + (long long)wh * TP * 64;
    const int total = T * NREL;
    for (int idx = blockIdx.x * 256 + threadIdx.x; idx < total;
         idx += gridDim.x * 256) {
        int qi = idx / NREL, j = idx % NREL;
        const float* qrow = qkv_row<WS, WPS>(qkv, qb, qi, w) + n * 64;
        const float* rh = relh + j * 64;
        const float* rw = relw + j * 64;
        float sh = 0.f, sw = 0.f;
#pragma unroll 8
        for (int c = 0; c < 64; c++) {
            float qv = qrow[c];
            sh += qv * rh[c];
            sw += qv * rw[c];
        }
        DhP[qi * 64 + j] = sh;
        DwP[qi * 64 + j] = sw;
    }
}

// ---------------------------------------------------------------------------
// Scores: S[q,k] = scale*(q.k) + Dh[q, qy-ky+S-1] + Dw[q, qx-kx+S-1]
// 256-thread block computes a 128x128 tile per (window,head): Q/K tiles
// (128 rows x 64 head dims, bf16) staged once in LDS, 16 wmma per wave.
// ---------------------------------------------------------------------------
template<bool GLOB>
__global__ void __launch_bounds__(256)
attn_score_kernel(const float* __restrict__ qkv, const float* __restrict__ qb,
                  const float* __restrict__ Dh, const float* __restrict__ Dw,
                  float* __restrict__ S, float scale)
{
    constexpr int TP  = GLOB ? 1024 : 256;
    constexpr int WS  = GLOB ? 32 : 14;
    constexpr int WPS = GLOB ? 1 : 3;
    constexpr int SR  = GLOB ? 32 : 14;
    constexpr int nmax = 2 * SR - 2;
    __shared__ __align__(16) unsigned short lQ[128 * 68];
    __shared__ __align__(16) unsigned short lK[128 * 68];
    const int tid  = threadIdx.x;
    const int lane = tid & 31;
    const int wave = tid >> 5;
    const int wm = wave >> 2, wn = wave & 3;
    const int q0 = blockIdx.x * 128;
    const int k0 = blockIdx.y * 128;
    const int wh = blockIdx.z;
    const int w = wh / 12, n = wh % 12;

    // Stage Q and K tiles: coalesced float4 loads, window padding resolved
    // once per row via qkv_row, single b64 LDS stores.
#pragma unroll
    for (int i = 0; i < 8; i++) {
        int idx = i * 256 + tid;               // 0..2047
        int r = idx >> 4, c4 = (idx & 15) * 4;
        const float* qr = qkv_row<WS, WPS>(qkv, qb, q0 + r, w) + n * 64 + c4;
        *(uint2*)&lQ[r * 68 + c4] = pack4(*(const float4*)qr);
        const float* kr = qkv_row<WS, WPS>(qkv, qb, k0 + r, w) + 768 + n * 64 + c4;
        *(uint2*)&lK[r * 68 + c4] = pack4(*(const float4*)kr);
    }
    __syncthreads();

    const v8f vzero = {0.f, 0.f, 0.f, 0.f, 0.f, 0.f, 0.f, 0.f};
    v8f acc[4][2];
#pragma unroll
    for (int a = 0; a < 4; a++)
#pragma unroll
        for (int b2 = 0; b2 < 2; b2++) acc[a][b2] = vzero;

#pragma unroll
    for (int kc = 0; kc < 64; kc += 32) {
#pragma unroll
        for (int mi = 0; mi < 4; mi++) {
            v16bf af = frag_a_ld(lQ, wm * 64 + mi * 16, 68, kc);
#pragma unroll
            for (int ni = 0; ni < 2; ni++) {
                v16bf bfv = frag_b_ld(lK, wn * 32 + ni * 16, 68, kc);
                acc[mi][ni] = __builtin_amdgcn_wmma_f32_16x16x32_bf16(
                    false, af, false, bfv, (short)0, acc[mi][ni], false, false);
            }
        }
    }

    const float* DhP = Dh + (long long)wh * TP * 64;
    const float* DwP = Dw + (long long)wh * TP * 64;
#pragma unroll
    for (int mi = 0; mi < 4; mi++) {
#pragma unroll
        for (int ni = 0; ni < 2; ni++) {
            int key = k0 + wn * 32 + ni * 16 + (lane & 15);
            int ky = key / WS, kx = key % WS;
            int qbase = q0 + wm * 64 + mi * 16 + (lane >> 4) * 8;
#pragma unroll
            for (int r = 0; r < 8; r++) {
                int qi = qbase + r;
                int qy = qi / WS, qx = qi % WS;
                int jh = qy - ky + SR - 1; jh = jh < 0 ? 0 : (jh > nmax ? nmax : jh);
                int jw = qx - kx + SR - 1; jw = jw < 0 ? 0 : (jw > nmax ? nmax : jw);
                S[((long long)wh * TP + qi) * TP + key] =
                    scale * acc[mi][ni][r] + DhP[(long long)qi * 64 + jh]
                                           + DwP[(long long)qi * 64 + jw];
            }
        }
    }
}

// ---------------------------------------------------------------------------
// Row softmax over first T cols; zero pad cols T..Tpad (so padded-K WMMA in
// the A*V stage contributes nothing).
// ---------------------------------------------------------------------------
__global__ void __launch_bounds__(256)
softmax_kernel(float* __restrict__ S, int T, int Tpad)
{
    __shared__ float red[256];
    const int row = blockIdx.x;
    const int wh  = blockIdx.y;
    float* p = S + ((long long)wh * Tpad + row) * Tpad;
    int tid = threadIdx.x;
    float m = -3.4e38f;
    for (int c = tid; c < T; c += 256) m = fmaxf(m, p[c]);
    red[tid] = m; __syncthreads();
    for (int s = 128; s > 0; s >>= 1) {
        if (tid < s) red[tid] = fmaxf(red[tid], red[tid + s]);
        __syncthreads();
    }
    m = red[0]; __syncthreads();
    float sum = 0.f;
    for (int c = tid; c < T; c += 256) { float e = __expf(p[c] - m); p[c] = e; sum += e; }
    red[tid] = sum; __syncthreads();
    for (int s = 128; s > 0; s >>= 1) {
        if (tid < s) red[tid] += red[tid + s];
        __syncthreads();
    }
    float inv = 1.f / red[0];
    for (int c = tid; c < T; c += 256) p[c] *= inv;
    for (int c = T + tid; c < Tpad; c += 256) p[c] = 0.f;
}

// ---------------------------------------------------------------------------
// O = P @ V: 256-thread block computes 128 queries x 64 head-dims per
// (window,head). Per 32-key step: stage P (128x32) + V^T (64x32) in LDS,
// 4 wmma per wave. Output written un-partitioned (token-major).
// ---------------------------------------------------------------------------
template<bool GLOB>
__global__ void __launch_bounds__(256)
attn_av_kernel(const float* __restrict__ S, const float* __restrict__ qkv,
               const float* __restrict__ qb, float* __restrict__ aout)
{
    constexpr int T   = GLOB ? 1024 : 196;
    constexpr int TP  = GLOB ? 1024 : 256;
    constexpr int WS  = GLOB ? 32 : 14;
    constexpr int WPS = GLOB ? 1 : 3;
    __shared__ __align__(16) unsigned short lP[128 * 36];
    __shared__ __align__(16) unsigned short lV[64 * 36];
    const int tid  = threadIdx.x;
    const int lane = tid & 31;
    const int wave = tid >> 5;
    const int wm = wave >> 1;          // 0..3 (M dir, 32 rows each)
    const int wn = wave & 1;           // 0..1 (N dir, 32 cols each)
    const int m0 = blockIdx.x * 128;
    const int wh = blockIdx.y;
    const int w = wh / 12, n = wh % 12;
    const float* Sw = S + (long long)wh * TP * TP;

    const v8f vzero = {0.f, 0.f, 0.f, 0.f, 0.f, 0.f, 0.f, 0.f};
    v8f acc[2][2];
#pragma unroll
    for (int a = 0; a < 2; a++)
#pragma unroll
        for (int b2 = 0; b2 < 2; b2++) acc[a][b2] = vzero;

    for (int kc = 0; kc < TP; kc += 32) {
        // stage P tile [128 q][32 k]
#pragma unroll
        for (int i = 0; i < 4; i++) {
            int idx = i * 256 + tid;
            int r = idx >> 3, c4 = (idx & 7) * 4;
            float4 v = *(const float4*)(Sw + (long long)(m0 + r) * TP + kc + c4);
            *(uint2*)&lP[r * 36 + c4] = pack4(v);
        }
        // stage V^T tile [64 c][32 k]
#pragma unroll
        for (int i = 0; i < 2; i++) {
            int idx = i * 256 + tid;           // 0..511
            int key = idx >> 4, c4 = (idx & 15) * 4;
            const float* vr =
                qkv_row<WS, WPS>(qkv, qb, kc + key, w) + 1536 + n * 64 + c4;
            float4 v = *(const float4*)vr;
            lV[(c4 + 0) * 36 + key] = f2bf(v.x);
            lV[(c4 + 1) * 36 + key] = f2bf(v.y);
            lV[(c4 + 2) * 36 + key] = f2bf(v.z);
            lV[(c4 + 3) * 36 + key] = f2bf(v.w);
        }
        __syncthreads();
#pragma unroll
        for (int mi = 0; mi < 2; mi++) {
            v16bf af = frag_a_ld(lP, wm * 32 + mi * 16, 36, 0);
#pragma unroll
            for (int ni = 0; ni < 2; ni++) {
                v16bf bfv = frag_b_ld(lV, wn * 32 + ni * 16, 36, 0);
                acc[mi][ni] = __builtin_amdgcn_wmma_f32_16x16x32_bf16(
                    false, af, false, bfv, (short)0, acc[mi][ni], false, false);
            }
        }
        __syncthreads();
    }

    int wy = w / WPS, wx = w % WPS;
#pragma unroll
    for (int mi = 0; mi < 2; mi++) {
#pragma unroll
        for (int ni = 0; ni < 2; ni++) {
            int c = wn * 32 + ni * 16 + (lane & 15);
            int qbase = m0 + wm * 32 + mi * 16 + (lane >> 4) * 8;
#pragma unroll
            for (int r = 0; r < 8; r++) {
                int qi = qbase + r;
                if (qi < T) {
                    int iy = qi / WS, ix = qi % WS;
                    int y = wy * WS + iy, x = wx * WS + ix;
                    if (y < 32 && x < 32)
                        aout[(long long)(y * 32 + x) * 768 + n * 64 + c] = acc[mi][ni][r];
                }
            }
        }
    }
}

// ---------------------------------------------------------------------------
// LayerNorm over last dim D (also used for LayerNorm2d with D=channels).
// One wave per row; generic output strides (final neck LN writes NCHW).
// ---------------------------------------------------------------------------
__global__ void __launch_bounds__(256)
ln_kernel(const float* __restrict__ in, const float* __restrict__ g,
          const float* __restrict__ b, float* __restrict__ out,
          int D, int rows, long long ors, long long ocs)
{
    int wave = threadIdx.x >> 5, lane = threadIdx.x & 31;
    int row = blockIdx.x * 8 + wave;
    if (row >= rows) return;
    const float* p = in + (long long)row * D;
    float s = 0.f, s2 = 0.f;
    for (int c = lane; c < D; c += 32) { float v = p[c]; s += v; s2 += v * v; }
#pragma unroll
    for (int off = 16; off > 0; off >>= 1) {
        s  += __shfl_xor(s,  off, 32);
        s2 += __shfl_xor(s2, off, 32);
    }
    float u = s / (float)D;
    float var = s2 / (float)D - u * u;
    float inv = rsqrtf(var + 1e-6f);
    for (int c = lane; c < D; c += 32)
        out[row * ors + c * ocs] = (p[c] - u) * inv * g[c] + b[c];
}

// ---------------------------------------------------------------------------
// im2col for 16x16/stride16 patch embed:  P[1024][768], k = ch*256+py*16+px
// ---------------------------------------------------------------------------
__global__ void __launch_bounds__(256)
im2col_patch(const float* __restrict__ x, float* __restrict__ P)
{
    int idx = blockIdx.x * 256 + threadIdx.x;
    if (idx >= 1024 * 768) return;
    int t = idx / 768, k = idx % 768;
    int ch = k >> 8, rem = k & 255, py = rem >> 4, px = rem & 15;
    int ty = t >> 5, tx = t & 31;
    P[idx] = x[ch * 262144 + (ty * 16 + py) * 512 + tx * 16 + px];
}

// im2col for 3x3/pad1 neck conv: P2[1024][2304], k = c*9 + dy*3 + dx
__global__ void __launch_bounds__(256)
im2col_3x3(const float* __restrict__ in, float* __restrict__ P2)
{
    int idx = blockIdx.x * 256 + threadIdx.x;
    if (idx >= 1024 * 2304) return;
    int t = idx / 2304, k = idx % 2304;
    int c = k / 9, r = k % 9, dy = r / 3, dx = r % 3;
    int y = t / 32 + dy - 1, x = t % 32 + dx - 1;
    P2[idx] = (y >= 0 && y < 32 && x >= 0 && x < 32)
                  ? in[(y * 32 + x) * 256 + c] : 0.f;
}

// ---------------------------------------------------------------------------
// Host orchestration
// ---------------------------------------------------------------------------
extern "C" void kernel_launch(void* const* d_in, const int* in_sizes, int n_in,
                              void* d_out, int out_size, void* d_ws, size_t ws_size,
                              hipStream_t stream)
{
    (void)in_sizes; (void)n_in; (void)out_size; (void)ws_size;
    const float* x         = (const float*)d_in[0];
    const float* patch_w   = (const float*)d_in[1];
    const float* patch_b   = (const float*)d_in[2];
    const float* pos_embed = (const float*)d_in[3];
    const float* ln1_g     = (const float*)d_in[4];
    const float* ln1_b     = (const float*)d_in[5];
    const float* qkv_w     = (const float*)d_in[6];
    const float* qkv_b     = (const float*)d_in[7];
    const float* proj_w    = (const float*)d_in[8];
    const float* proj_b    = (const float*)d_in[9];
    const float* rel_h     = (const float*)d_in[10];
    const float* rel_w     = (const float*)d_in[11];
    const float* ln2_g     = (const float*)d_in[12];
    const float* ln2_b     = (const float*)d_in[13];
    const float* fc1_w     = (const float*)d_in[14];
    const float* fc1_b     = (const float*)d_in[15];
    const float* fc2_w     = (const float*)d_in[16];
    const float* fc2_b     = (const float*)d_in[17];
    const float* neck_w1   = (const float*)d_in[18];
    const float* nln1_g    = (const float*)d_in[19];
    const float* nln1_b    = (const float*)d_in[20];
    const float* neck_w2   = (const float*)d_in[21];
    const float* nln2_g    = (const float*)d_in[22];
    const float* nln2_b    = (const float*)d_in[23];
    float* out = (float*)d_out;

    // Workspace carve-up (~100 MB), fully rewritten each call.
    float* W = (float*)d_ws;
    size_t off = 0;
    float* h    = W + off; off += (size_t)1024 * 768;
    float* xn   = W + off; off += (size_t)1024 * 768;
    float* qkvb = W + off; off += (size_t)1024 * 2304;   // also neck im2col
    float* aout = W + off; off += (size_t)1024 * 768;    // also neck conv1 out
    float* mbuf = W + off; off += (size_t)1024 * 3072;   // also patch im2col / neck conv2 out
    float* Sbuf = W + off; off += (size_t)12 * 1024 * 1024;   // >= 108*256*256
    float* DhB  = W + off; off += (size_t)108 * 256 * 64;
    float* DwB  = W + off; off += (size_t)108 * 256 * 64;

    dim3 b256(256);

    // ---- patch embed: im2col + GEMM (+bias, +pos_embed residual) -> h
    im2col_patch<<<(1024 * 768 + 255) / 256, b256, 0, stream>>>(x, mbuf);
    gemm_kernel<true, true, false><<<dim3(6, 8), b256, 0, stream>>>(
        mbuf, 768, patch_w, 1, 768, patch_b, pos_embed, h, 1024, 768, 768);

    const float scale = 0.125f;   // 64^-0.5
    for (int i = 0; i < 12; i++) {
        bool isglob = (i == 2 || i == 5 || i == 8 || i == 11);
        int T    = isglob ? 1024 : 196;
        int Tpad = isglob ? 1024 : 256;   // multiple of 128 for tile shapes
        int Srel = isglob ? 32 : 14;
        int nwin = isglob ? 1 : 9;
        const float* qb = qkv_b + (size_t)i * 2304;

        ln_kernel<<<128, b256, 0, stream>>>(h, ln1_g + i * 768, ln1_b + i * 768,
                                            xn, 768, 1024, 768, 1);
        gemm_kernel<true, false, false><<<dim3(18, 8), b256, 0, stream>>>(
            xn, 768, qkv_w + (size_t)i * 768 * 2304, 2304, 1, qb, nullptr,
            qkvb, 1024, 2304, 768);

        int total = T * (2 * Srel - 1);
        dim3 relg((total + 255) / 256, nwin * 12);
        dim3 scg(Tpad / 128, Tpad / 128, nwin * 12);
        dim3 smg(T, nwin * 12);
        dim3 avg(Tpad / 128, nwin * 12);
        const float* rhp = rel_h + (size_t)i * 63 * 64;
        const float* rwp = rel_w + (size_t)i * 63 * 64;
        if (isglob) {
            relq_kernel<true><<<relg, b256, 0, stream>>>(qkvb, qb, rhp, rwp, DhB, DwB);
            attn_score_kernel<true><<<scg, b256, 0, stream>>>(qkvb, qb, DhB, DwB, Sbuf, scale);
            softmax_kernel<<<smg, b256, 0, stream>>>(Sbuf, T, Tpad);
            attn_av_kernel<true><<<avg, b256, 0, stream>>>(Sbuf, qkvb, qb, aout);
        } else {
            relq_kernel<false><<<relg, b256, 0, stream>>>(qkvb, qb, rhp, rwp, DhB, DwB);
            attn_score_kernel<false><<<scg, b256, 0, stream>>>(qkvb, qb, DhB, DwB, Sbuf, scale);
            softmax_kernel<<<smg, b256, 0, stream>>>(Sbuf, T, Tpad);
            attn_av_kernel<false><<<avg, b256, 0, stream>>>(Sbuf, qkvb, qb, aout);
        }

        gemm_kernel<true, true, false><<<dim3(6, 8), b256, 0, stream>>>(
            aout, 768, proj_w + (size_t)i * 768 * 768, 768, 1,
            proj_b + i * 768, h, h, 1024, 768, 768);

        ln_kernel<<<128, b256, 0, stream>>>(h, ln2_g + i * 768, ln2_b + i * 768,
                                            xn, 768, 1024, 768, 1);

        gemm_kernel<true, false, true><<<dim3(24, 8), b256, 0, stream>>>(
            xn, 768, fc1_w + (size_t)i * 768 * 3072, 3072, 1,
            fc1_b + i * 3072, nullptr, mbuf, 1024, 3072, 768);

        gemm_kernel<true, true, false><<<dim3(6, 8), b256, 0, stream>>>(
            mbuf, 3072, fc2_w + (size_t)i * 3072 * 768, 768, 1,
            fc2_b + i * 768, h, h, 1024, 768, 3072);
    }

    // ---- neck
    gemm_kernel<false, false, false><<<dim3(2, 8), b256, 0, stream>>>(
        h, 768, neck_w1, 1, 768, nullptr, nullptr, aout, 1024, 256, 768);
    ln_kernel<<<128, b256, 0, stream>>>(aout, nln1_g, nln1_b, xn, 256, 1024, 256, 1);
    im2col_3x3<<<(1024 * 2304 + 255) / 256, b256, 0, stream>>>(xn, qkvb);
    gemm_kernel<false, false, false><<<dim3(2, 8), b256, 0, stream>>>(
        qkvb, 2304, neck_w2, 1, 2304, nullptr, nullptr, mbuf, 1024, 256, 2304);
    // final LayerNorm2d writing NCHW [256,32,32] directly to d_out
    ln_kernel<<<128, b256, 0, stream>>>(mbuf, nln2_g, nln2_b, out, 256, 1024, 1, 1024);
}
